// MutualMultiAttentionHead_34428457845176
// MI455X (gfx1250) — matrix operand order repaired and checked by reading, not compiled, not run on hardware
//
#include <hip/hip_runtime.h>

typedef __attribute__((ext_vector_type(16))) _Float16 v16h;
typedef __attribute__((ext_vector_type(8)))  float    v8f;

#define NN   3072
#define EE   3072
#define ELG  6144
#define FD   64
#define NH   4
#define NC   256   // H*F, fixed output width of the q/v projections

// ---------------------------------------------------------------------------
// WMMA GEMM specialized for out[M,256] = X[M,64] @ W[64,256] + bias.
// Block = 256 threads (8 waves). The block stages all of W into LDS once,
// pre-swizzled into exact 16x16x32 B-fragment layout (per lane: 16 contiguous
// halves -> 2x ds_load_b128). Each wave owns one 16-row stripe of A (loaded
// once as float4, converted to f16) and sweeps all 16 column tiles:
// 16 tiles x 2 WMMAs = 32 v_wmma_f32_16x16x32_f16 per wave.
// Fragment layouts per CDNA5 ISA 7.12.2:
//   A 16x32: lanes 0-15 hold K {0..7,16..23}; lanes 16-31 hold {8..15,24..31}
//   B 32x16: lanes 0-15 hold K {0..7,16..23}; lanes 16-31 hold {8..15,24..31}
//   C/D:     VGPR r -> row r + 8*(lane>=16), col = lane&15
// ---------------------------------------------------------------------------
__global__ void __launch_bounds__(256)
gemm_wmma_f16(const float* __restrict__ X,
              const float* __restrict__ W,
              const float* __restrict__ bias,
              float* __restrict__ out, int M) {
    // [tn][kb][lane][i] : 16*2*32*16 halves = 32 KB
    __shared__ _Float16 sw[16 * 2 * 32 * 16];
    __shared__ float    sbias[NC];

    const int t = threadIdx.x;

    // Stage W swizzled; pos enumerates (k, col) with col fastest -> coalesced.
    for (int pos = t; pos < FD * NC; pos += 256) {
        const int k   = pos >> 8;        // 0..63
        const int col = pos & 255;       // 0..255
        const int tn  = col >> 4;
        const int kb  = k >> 5;
        const int l   = (((k >> 3) & 1) << 4) | (col & 15);
        const int i   = (k & 7) | (((k >> 4) & 1) << 3);
        sw[(((tn << 1) + kb) * 32 + l) * 16 + i] = (_Float16)W[pos];
    }
    sbias[t] = bias[t];
    __syncthreads();

    const int wave = t >> 5, lane = t & 31;
    const int tm   = blockIdx.x * 8 + wave;     // 16-row stripe owned by wave
    if (tm * 16 >= M) return;                   // wave-uniform (EXEC stays full)

    const int row  = tm * 16 + (lane & 15);
    const int hi   = lane >> 4;
    const int koff = hi * 8;

    // A fragments for K=0..63, loaded once via float4.
    const float4* x4 = (const float4*)(X + (size_t)row * FD);
    const int q = koff >> 2;                    // 0 or 2
    const float4 f0 = x4[q],      f1 = x4[q + 1];    // K  koff..koff+7
    const float4 f2 = x4[q + 4],  f3 = x4[q + 5];    // K  16+koff..
    const float4 f4 = x4[q + 8],  f5 = x4[q + 9];    // K  32+koff..
    const float4 f6 = x4[q + 12], f7 = x4[q + 13];   // K  48+koff..
    v16h a0{}, a1{};
    a0[0] = (_Float16)f0.x; a0[1] = (_Float16)f0.y; a0[2] = (_Float16)f0.z; a0[3] = (_Float16)f0.w;
    a0[4] = (_Float16)f1.x; a0[5] = (_Float16)f1.y; a0[6] = (_Float16)f1.z; a0[7] = (_Float16)f1.w;
    a0[8]  = (_Float16)f2.x; a0[9]  = (_Float16)f2.y; a0[10] = (_Float16)f2.z; a0[11] = (_Float16)f2.w;
    a0[12] = (_Float16)f3.x; a0[13] = (_Float16)f3.y; a0[14] = (_Float16)f3.z; a0[15] = (_Float16)f3.w;
    a1[0] = (_Float16)f4.x; a1[1] = (_Float16)f4.y; a1[2] = (_Float16)f4.z; a1[3] = (_Float16)f4.w;
    a1[4] = (_Float16)f5.x; a1[5] = (_Float16)f5.y; a1[6] = (_Float16)f5.z; a1[7] = (_Float16)f5.w;
    a1[8]  = (_Float16)f6.x; a1[9]  = (_Float16)f6.y; a1[10] = (_Float16)f6.z; a1[11] = (_Float16)f6.w;
    a1[12] = (_Float16)f7.x; a1[13] = (_Float16)f7.y; a1[14] = (_Float16)f7.z; a1[15] = (_Float16)f7.w;

    const int colLo = lane & 15;
    #pragma unroll
    for (int tn = 0; tn < 16; ++tn) {
        const v16h b0 = *(const v16h*)&sw[((tn * 2 + 0) * 32 + lane) * 16];
        const v16h b1 = *(const v16h*)&sw[((tn * 2 + 1) * 32 + lane) * 16];
        v8f c = {};
        c = __builtin_amdgcn_wmma_f32_16x16x32_f16(false, a0, false, b0, (short)0, c, false, false);
        c = __builtin_amdgcn_wmma_f32_16x16x32_f16(false, a1, false, b1, (short)0, c, false, false);
        const int col = tn * 16 + colLo;
        const float bv = sbias[col];
        #pragma unroll
        for (int r = 0; r < 8; ++r)
            out[(size_t)(tm * 16 + r + hi * 8) * NC + col] = c[r] + bv;
    }
}

// ---------------------------------------------------------------------------
// k-projection: out[M,4] = X[M,64] @ Wk[64,4] + bk  (too skinny for WMMA tile)
// ---------------------------------------------------------------------------
__global__ void proj_k(const float* __restrict__ X, const float* __restrict__ Wk,
                       const float* __restrict__ bk, float* __restrict__ out, int M) {
    int idx = blockIdx.x * blockDim.x + threadIdx.x;
    if (idx >= M * 4) return;
    int m = idx >> 2, cidx = idx & 3;
    float acc = bk[cidx];
    #pragma unroll 8
    for (int f = 0; f < FD; ++f) acc += X[(size_t)m * FD + f] * Wk[f * 4 + cidx];
    out[idx] = acc;
}

// ---------------------------------------------------------------------------
// srk[h,f] = sum_m SR[m,f] * K[h*M+m].  One block per head, 256 threads.
// ---------------------------------------------------------------------------
__global__ void srk_kernel(const float* __restrict__ SR, const float* __restrict__ Kf,
                           float* __restrict__ srk, int M) {
    const int h = blockIdx.x;
    const int t = threadIdx.x;
    const int f = t & 63, part = t >> 6;
    const float* kh = Kf + (size_t)h * M;
    float acc = 0.f;
    for (int m = part; m < M; m += 4) acc += SR[(size_t)m * FD + f] * kh[m];
    __shared__ float red[256];
    red[t] = acc;
    __syncthreads();
    if (part == 0) srk[h * FD + f] = red[f] + red[64 + f] + red[128 + f] + red[192 + f];
}

// ---------------------------------------------------------------------------
// att[h,m] = softmax_m( q[h,m,:].srk[h,:] / 8 ).  One block (1024 thr) per head.
// ---------------------------------------------------------------------------
__global__ void attn_softmax(const float* __restrict__ Q,   // [H*M, 64] flat view
                             const float* __restrict__ srk, // [H, 64]
                             float* __restrict__ att, int M) {
    const int h = blockIdx.x;
    const int t = threadIdx.x;
    __shared__ float xs[NN];
    __shared__ float red[1024];
    __shared__ float s_srk[FD];
    if (t < FD) s_srk[t] = srk[h * FD + t];
    __syncthreads();

    for (int m = t; m < M; m += 1024) {
        const float4* qr = (const float4*)(Q + ((size_t)h * M + m) * FD);
        float acc = 0.f;
        #pragma unroll
        for (int i = 0; i < 16; ++i) {
            float4 q4 = qr[i];
            acc += q4.x * s_srk[4*i] + q4.y * s_srk[4*i+1]
                 + q4.z * s_srk[4*i+2] + q4.w * s_srk[4*i+3];
        }
        xs[m] = acc * 0.125f;   // 1/sqrt(64)
    }
    __syncthreads();

    float mx = -1e30f;
    for (int m = t; m < M; m += 1024) mx = fmaxf(mx, xs[m]);
    red[t] = mx; __syncthreads();
    for (int s = 512; s > 0; s >>= 1) {
        if (t < s) red[t] = fmaxf(red[t], red[t + s]);
        __syncthreads();
    }
    mx = red[0];
    __syncthreads();

    float sum = 0.f;
    for (int m = t; m < M; m += 1024) { float e = __expf(xs[m] - mx); xs[m] = e; sum += e; }
    red[t] = sum; __syncthreads();
    for (int s = 512; s > 0; s >>= 1) {
        if (t < s) red[t] += red[t + s];
        __syncthreads();
    }
    const float inv = 1.f / red[0];
    for (int m = t; m < M; m += 1024) att[(size_t)h * M + m] = xs[m] * inv;
}

// ---------------------------------------------------------------------------
// node_msg (head-mean): out[src[e],f] += 0.25 * sum_h eatt[h,e]*nodeV[h*M+dst[e],f]
// ---------------------------------------------------------------------------
__global__ void scatter_node(const int* __restrict__ src, const int* __restrict__ dst,
                             const float* __restrict__ eatt, const float* __restrict__ nodeV,
                             float* __restrict__ out, int E_, int M) {
    const int e = blockIdx.x;
    const int f = threadIdx.x;
    const int s = src[e], d = dst[e];
    float acc = 0.f;
    #pragma unroll
    for (int h = 0; h < NH; ++h)
        acc += eatt[(size_t)h * E_ + e] * nodeV[((size_t)(h * M + d)) * FD + f];
    atomicAdd(&out[(size_t)s * FD + f], acc * 0.25f);
}

// ---------------------------------------------------------------------------
// edge_msg: out[lg_src[l],f] += 0.25*sum_h natt[h,dst[lg_src[l]]]*edgeV[h*E+lg_dst[l],f]
// ---------------------------------------------------------------------------
__global__ void scatter_edge(const int* __restrict__ lg_src, const int* __restrict__ lg_dst,
                             const int* __restrict__ dst, const float* __restrict__ natt,
                             const float* __restrict__ edgeV, float* __restrict__ out,
                             int Nn_, int Ee_) {
    const int l  = blockIdx.x;
    const int f  = threadIdx.x;
    const int e1 = lg_src[l], e2 = lg_dst[l];
    const int conn = dst[e1];
    float acc = 0.f;
    #pragma unroll
    for (int h = 0; h < NH; ++h)
        acc += natt[(size_t)h * Nn_ + conn] * edgeV[((size_t)(h * Ee_ + e2)) * FD + f];
    atomicAdd(&out[(size_t)e1 * FD + f], acc * 0.25f);
}

__global__ void zero_kernel(float* __restrict__ p, int n) {
    int i = blockIdx.x * blockDim.x + threadIdx.x;
    if (i < n) p[i] = 0.f;
}

// ---------------------------------------------------------------------------
extern "C" void kernel_launch(void* const* d_in, const int* in_sizes, int n_in,
                              void* d_out, int out_size, void* d_ws, size_t ws_size,
                              hipStream_t stream) {
    const float* node_in = (const float*)d_in[0];
    const float* edge_in = (const float*)d_in[1];
    const int*   src     = (const int*)d_in[2];
    const int*   dst     = (const int*)d_in[3];
    const int*   lg_src  = (const int*)d_in[4];
    const int*   lg_dst  = (const int*)d_in[5];
    const float* nqW = (const float*)d_in[6],  *nqb = (const float*)d_in[7];
    const float* nkW = (const float*)d_in[8],  *nkb = (const float*)d_in[9];
    const float* nvW = (const float*)d_in[10], *nvb = (const float*)d_in[11];
    const float* eqW = (const float*)d_in[12], *eqb = (const float*)d_in[13];
    const float* ekW = (const float*)d_in[14], *ekb = (const float*)d_in[15];
    const float* evW = (const float*)d_in[16], *evb = (const float*)d_in[17];

    // workspace carve-out (floats): ~12.7 MB total
    float* ws    = (float*)d_ws;
    float* nodeQ = ws; ws += (size_t)NN * NC;   // [N,256] == q flat [H*N,64]
    float* nodeV = ws; ws += (size_t)NN * NC;
    float* nodeK = ws; ws += (size_t)NN * NH;   // [N,4] flat == k flat [H*N]
    float* edgeQ = ws; ws += (size_t)EE * NC;
    float* edgeV = ws; ws += (size_t)EE * NC;
    float* edgeK = ws; ws += (size_t)EE * NH;
    float* srkN  = ws; ws += NH * FD;
    float* srkE  = ws; ws += NH * FD;
    float* attN  = ws; ws += (size_t)NH * NN;
    float* attE  = ws; ws += (size_t)NH * EE;

    // Projections: 24 blocks x 8 waves, each wave = 16-row stripe x all 256 cols
    const dim3 gB(NN / 128), gT(256);
    gemm_wmma_f16<<<gB, gT, 0, stream>>>(node_in, nqW, nqb, nodeQ, NN);
    gemm_wmma_f16<<<gB, gT, 0, stream>>>(node_in, nvW, nvb, nodeV, NN);
    gemm_wmma_f16<<<gB, gT, 0, stream>>>(edge_in, eqW, eqb, edgeQ, EE);
    gemm_wmma_f16<<<gB, gT, 0, stream>>>(edge_in, evW, evb, edgeV, EE);
    proj_k<<<(NN * 4 + 255) / 256, 256, 0, stream>>>(node_in, nkW, nkb, nodeK, NN);
    proj_k<<<(EE * 4 + 255) / 256, 256, 0, stream>>>(edge_in, ekW, ekb, edgeK, EE);

    // Attention scores
    srk_kernel<<<NH, 256, 0, stream>>>(node_in, nodeK, srkN, NN);
    srk_kernel<<<NH, 256, 0, stream>>>(edge_in, edgeK, srkE, EE);
    attn_softmax<<<NH, 1024, 0, stream>>>(nodeQ, srkN, attN, NN);
    attn_softmax<<<NH, 1024, 0, stream>>>(edgeQ, srkE, attE, EE);

    // Sparse message passing + head mean into d_out
    float* out_node = (float*)d_out;
    float* out_edge = out_node + (size_t)NN * FD;
    zero_kernel<<<(out_size + 255) / 256, 256, 0, stream>>>((float*)d_out, out_size);
    scatter_node<<<EE, FD, 0, stream>>>(src, dst, attE, nodeV, out_node, EE, NN);
    scatter_edge<<<ELG, FD, 0, stream>>>(lg_src, lg_dst, dst, attN, edgeV, out_edge, NN, EE);
}
